// Res5ROIHeads_63866163692128
// MI455X (gfx1250) — matrix operand order repaired and checked by reading, not compiled, not run on hardware
//
#include <hip/hip_runtime.h>

// ---------------------------------------------------------------------------
// Types
// ---------------------------------------------------------------------------
typedef __bf16 bf16_t;
typedef __attribute__((ext_vector_type(16))) __bf16 v16bf;
typedef __attribute__((ext_vector_type(8)))  __bf16 v8bf;
typedef __attribute__((ext_vector_type(8)))  float  v8f;

__device__ __forceinline__ bf16_t u16_as_bf(unsigned short u) {
    return __builtin_bit_cast(bf16_t, u);
}
__device__ __forceinline__ bf16_t f2bf(float f) {   // round-to-nearest-even
    unsigned u = __builtin_bit_cast(unsigned, f);
    u += 0x7fffu + ((u >> 16) & 1u);
    return u16_as_bf((unsigned short)(u >> 16));
}
__device__ __forceinline__ float bf2f(bf16_t h) {
    unsigned u = (unsigned)__builtin_bit_cast(unsigned short, h) << 16;
    return __builtin_bit_cast(float, u);
}

// ---------------------------------------------------------------------------
// Problem constants
// ---------------------------------------------------------------------------
#define NBOX   128
#define CIN    1024
#define CMID   512
#define COUT   2048
#define PP     14
#define M0     (NBOX*PP*PP)  // 25088
#define M1     (NBOX*7*7)    // 6272
#define NCLS   1601
#define NBBX   6400

enum { AM_PLAIN = 0, AM_STRIDE2 = 1, AM_IM2COL = 2 };
enum { EP_BNRELU_BF16 = 0, EP_BN_F32 = 1, EP_BN_ADDF32_RELU_BF16 = 2,
       EP_BN_ADDBF16_RELU_BF16 = 3, EP_BIAS_F32 = 4 };

#define LDSROW     80                  // 64B of data + 16B pad (bank spread)
#define TILEBYTES  (128 * LDSROW)      // 10240 B per tile buffer

__device__ __forceinline__ int srow_stride2(int m) {
    // m = r*49 + i*7 + j -> source row r*196 + 2i*14 + 2j (14x14 input)
    const int r = m / 49; const int t = m - r * 49;
    const int i = t / 7;  const int j = t - i * 7;
    return r * 196 + i * 28 + j * 2;
}

__device__ __forceinline__ unsigned long long im2col_src(
    const bf16_t* __restrict__ A, const bf16_t* __restrict__ zbuf,
    int r, int oy, int ox, int kk, int Hin, int stride)
{
    const int f  = kk >> 9;          // (ky*3+kx), 0..8
    const int ci = kk & 511;
    const int ky = f / 3, kx = f - ky * 3;
    const int iy = oy * stride - 1 + ky;
    const int ix = ox * stride - 1 + kx;
    const bf16_t* p = (iy >= 0 && iy < Hin && ix >= 0 && ix < Hin)
        ? A + ((size_t)((r * Hin + iy) * Hin + ix)) * 512 + ci
        : zbuf;
    return (unsigned long long)(uintptr_t)p;
}

// ---------------------------------------------------------------------------
// Generic WMMA GEMM with async-LDS double buffering.
//   out[m,n] = epilogue( sum_k A[m,k] * Bw[n,k] )
// Block: 256 threads = 8 waves; block tile 128x128; wave tile 64x32.
// Grid: (ceil(N/128), M/128). M must be a multiple of 128 (true for all uses).
// K multiple of 32.
// ---------------------------------------------------------------------------
template<int AMODE, int EPI>
__global__ __launch_bounds__(256) void k_gemm(
    const bf16_t* __restrict__ A, const bf16_t* __restrict__ Bw,
    const float* __restrict__ scale, const float* __restrict__ bias,
    const void* __restrict__ res, void* __restrict__ out,
    const bf16_t* __restrict__ zbuf,
    int N, int K, int ldo, int Hin, int stride)
{
    __shared__ alignas(16) unsigned char sA[2 * TILEBYTES];
    __shared__ alignas(16) unsigned char sB[2 * TILEBYTES];

    const int tid   = threadIdx.x;
    const int lane  = tid & 31;
    const int wave  = tid >> 5;
    const int mBlock = blockIdx.y * 128;
    const int nBlock = blockIdx.x * 128;
    const int half  = lane >> 4;
    const int lr    = lane & 15;

    // ---- staging setup: thread covers chunks (rc, kc) and (rc+64, kc) ----
    const int rc = tid >> 2;                 // 0..63
    const int kc = tid & 3;                  // 16B chunk within 64B k-row
    const unsigned baseA = (unsigned)(uintptr_t)&sA[0];
    const unsigned baseB = (unsigned)(uintptr_t)&sB[0];
    const unsigned dA0 = baseA + rc * LDSROW + kc * 16;
    const unsigned dA1 = baseA + (rc + 64) * LDSROW + kc * 16;
    const unsigned dB0 = baseB + rc * LDSROW + kc * 16;
    const unsigned dB1 = baseB + (rc + 64) * LDSROW + kc * 16;

    const int mA0 = mBlock + rc, mA1 = mBlock + rc + 64;
    const bf16_t *srcA0 = nullptr, *srcA1 = nullptr;
    int r0_ = 0, oy0_ = 0, ox0_ = 0, r1_ = 0, oy1_ = 0, ox1_ = 0;
    if (AMODE == AM_PLAIN) {
        srcA0 = A + (size_t)mA0 * K + kc * 8;
        srcA1 = A + (size_t)mA1 * K + kc * 8;
    } else if (AMODE == AM_STRIDE2) {
        srcA0 = A + (size_t)srow_stride2(mA0) * K + kc * 8;
        srcA1 = A + (size_t)srow_stride2(mA1) * K + kc * 8;
    } else {
        r0_ = mA0 / 49; { int t = mA0 - r0_ * 49; oy0_ = t / 7; ox0_ = t - oy0_ * 7; }
        r1_ = mA1 / 49; { int t = mA1 - r1_ * 49; oy1_ = t / 7; ox1_ = t - oy1_ * 7; }
    }
    const int nB0 = min(nBlock + rc, N - 1);
    const int nB1 = min(nBlock + rc + 64, N - 1);
    const bf16_t* srcB0 = Bw + (size_t)nB0 * K + kc * 8;
    const bf16_t* srcB1 = Bw + (size_t)nB1 * K + kc * 8;

    const int T = K >> 5;   // number of 32-wide k tiles

    auto issue = [&](int kt) {
        const unsigned bo = (unsigned)(kt & 1) * TILEBYTES;
        unsigned long long a0, a1;
        if (AMODE == AM_IM2COL) {
            const int kk = kt * 32 + kc * 8;
            a0 = im2col_src(A, zbuf, r0_, oy0_, ox0_, kk, Hin, stride);
            a1 = im2col_src(A, zbuf, r1_, oy1_, ox1_, kk, Hin, stride);
        } else {
            a0 = (unsigned long long)(uintptr_t)srcA0; srcA0 += 32;
            a1 = (unsigned long long)(uintptr_t)srcA1; srcA1 += 32;
        }
        unsigned long long b0 = (unsigned long long)(uintptr_t)srcB0; srcB0 += 32;
        unsigned long long b1 = (unsigned long long)(uintptr_t)srcB1; srcB1 += 32;
        asm volatile("global_load_async_to_lds_b128 %0, %1, off"
                     :: "v"(dA0 + bo), "v"(a0) : "memory");
        asm volatile("global_load_async_to_lds_b128 %0, %1, off"
                     :: "v"(dA1 + bo), "v"(a1) : "memory");
        asm volatile("global_load_async_to_lds_b128 %0, %1, off"
                     :: "v"(dB0 + bo), "v"(b0) : "memory");
        asm volatile("global_load_async_to_lds_b128 %0, %1, off"
                     :: "v"(dB1 + bo), "v"(b1) : "memory");
    };

    // ---- accumulators ----
    v8f acc[4][2];
    #pragma unroll
    for (int i = 0; i < 4; ++i)
        #pragma unroll
        for (int j = 0; j < 2; ++j)
            #pragma unroll
            for (int v = 0; v < 8; ++v) acc[i][j][v] = 0.0f;

    // ---- pipeline ----
    issue(0);
    for (int kt = 0; kt < T; ++kt) {
        if (kt + 1 < T) {
            issue(kt + 1);
            // 4 outstanding = tile kt complete (async loads complete in order)
            asm volatile("s_wait_asynccnt 0x4" ::: "memory");
        } else {
            asm volatile("s_wait_asynccnt 0x0" ::: "memory");
        }
        __syncthreads();   // tile kt visible to all waves

        const unsigned char* bufA = sA + (kt & 1) * TILEBYTES;
        const unsigned char* bufB = sB + (kt & 1) * TILEBYTES;

        v16bf a[4];
        #pragma unroll
        for (int i = 0; i < 4; ++i) {
            // A frag: lane -> row (wave&1)*64 + 16*i + lr, K chunks half*8(+16)
            const unsigned char* p =
                bufA + ((wave & 1) * 64 + 16 * i + lr) * LDSROW + half * 16;
            v8bf* a8 = reinterpret_cast<v8bf*>(&a[i]);
            a8[0] = *reinterpret_cast<const v8bf*>(p);
            a8[1] = *reinterpret_cast<const v8bf*>(p + 32);
        }
        v16bf b[2];
        #pragma unroll
        for (int j = 0; j < 2; ++j) {
            // B frag: lane -> col (wave>>1)*32 + 16*j + lr, K chunks half*8(+16)
            const unsigned char* p =
                bufB + ((wave >> 1) * 32 + 16 * j + lr) * LDSROW + half * 16;
            v8bf* b8 = reinterpret_cast<v8bf*>(&b[j]);
            b8[0] = *reinterpret_cast<const v8bf*>(p);
            b8[1] = *reinterpret_cast<const v8bf*>(p + 32);
        }
        #pragma unroll
        for (int i = 0; i < 4; ++i)
            #pragma unroll
            for (int j = 0; j < 2; ++j)
                acc[i][j] = __builtin_amdgcn_wmma_f32_16x16x32_bf16(
                    false, a[i], false, b[j], (short)0, acc[i][j], false, false);

        __syncthreads();   // reads done before buffer (kt&1) is refilled
    }

    // ---- epilogue: lane -> n = lr; accum elem v -> m = 16*i + v + 8*half ----
    const int mWave = mBlock + (wave & 1) * 64;
    const int nWave = nBlock + (wave >> 1) * 32;
    #pragma unroll
    for (int j = 0; j < 2; ++j) {
        const int nn = nWave + 16 * j + lr;
        if (nn >= N) continue;
        const float sc = (EPI == EP_BIAS_F32) ? 1.0f : scale[nn];
        const float bs = bias[nn];
        #pragma unroll
        for (int i = 0; i < 4; ++i) {
            #pragma unroll
            for (int v = 0; v < 8; ++v) {
                const int mm = mWave + 16 * i + v + 8 * half;
                float y = acc[i][j][v] * sc + bs;
                const size_t o = (size_t)mm * ldo + nn;
                if (EPI == EP_BNRELU_BF16) {
                    y = y > 0.0f ? y : 0.0f;
                    ((bf16_t*)out)[o] = f2bf(y);
                } else if (EPI == EP_BN_F32) {
                    ((float*)out)[o] = y;
                } else if (EPI == EP_BN_ADDF32_RELU_BF16) {
                    y += ((const float*)res)[o];
                    y = y > 0.0f ? y : 0.0f;
                    ((bf16_t*)out)[o] = f2bf(y);
                } else if (EPI == EP_BN_ADDBF16_RELU_BF16) {
                    y += bf2f(((const bf16_t*)res)[o]);
                    y = y > 0.0f ? y : 0.0f;
                    ((bf16_t*)out)[o] = f2bf(y);
                } else { // EP_BIAS_F32
                    ((float*)out)[o] = y;
                }
            }
        }
    }
}

// ---------------------------------------------------------------------------
// ROI Align -> X0 bf16 [25088, 1024]  (row = r*196 + py*14 + px, col = channel)
// ---------------------------------------------------------------------------
__global__ void k_roi(const float* __restrict__ feat,
                      const float* __restrict__ boxes,
                      const int* __restrict__ bidx,
                      bf16_t* __restrict__ X0)
{
    const int idx = blockIdx.x;
    const int r  = idx / 196;
    const int t  = idx - r * 196;
    const int py = t / 14, px = t - py * 14;
    const float s = 1.0f / 16.0f;
    const float bx1 = boxes[r * 4 + 0] * s - 0.5f;
    const float by1 = boxes[r * 4 + 1] * s - 0.5f;
    const float bx2 = boxes[r * 4 + 2] * s - 0.5f;
    const float by2 = boxes[r * 4 + 3] * s - 0.5f;
    const float cy = ((float)py + 0.5f) / 14.0f;
    const float cx = ((float)px + 0.5f) / 14.0f;
    const float ys = by1 + cy * (by2 - by1);
    const float xs = bx1 + cx * (bx2 - bx1);
    const bool valid = (ys >= -1.0f) && (ys <= 64.0f) &&
                       (xs >= -1.0f) && (xs <= 64.0f);
    const float yc = fminf(fmaxf(ys, 0.0f), 63.0f);
    const float xc = fminf(fmaxf(xs, 0.0f), 63.0f);
    const int y0 = (int)floorf(yc), x0 = (int)floorf(xc);
    const int y1 = min(y0 + 1, 63), x1 = min(x0 + 1, 63);
    const float ly = yc - (float)y0, lx = xc - (float)x0;
    const float hy = 1.0f - ly, hx = 1.0f - lx;
    const float w00 = hy * hx, w01 = hy * lx, w10 = ly * hx, w11 = ly * lx;
    const int bi = bidx[r];
    const float* fb = feat + (size_t)bi * 1024 * 4096;
    const int o00 = y0 * 64 + x0, o01 = y0 * 64 + x1;
    const int o10 = y1 * 64 + x0, o11 = y1 * 64 + x1;
    for (int c = threadIdx.x; c < 1024; c += blockDim.x) {
        const float* fc = fb + (size_t)c * 4096;
        float v = w00 * fc[o00] + w01 * fc[o01] + w10 * fc[o10] + w11 * fc[o11];
        X0[(size_t)idx * 1024 + c] = f2bf(valid ? v : 0.0f);
    }
}

// ---------------------------------------------------------------------------
// Weight conversion / zero-fill kernels
// ---------------------------------------------------------------------------
__global__ void k_convert(const float* __restrict__ s, bf16_t* __restrict__ d,
                          int n)
{
    const int i = blockIdx.x * blockDim.x + threadIdx.x;
    if (i < n) d[i] = f2bf(s[i]);
}

// OIHW (512,512,3,3) -> [n][ (ky*3+kx)*512 + ci ] bf16
__global__ void k_convert3x3(const float* __restrict__ s, bf16_t* __restrict__ d)
{
    const int i = blockIdx.x * blockDim.x + threadIdx.x;
    if (i >= 512 * 4608) return;
    const int n = i / 4608;
    const int rrem = i - n * 4608;
    const int f = rrem >> 9;
    const int ci = rrem & 511;
    d[i] = f2bf(s[(size_t)n * 4608 + ci * 9 + f]);
}

__global__ void k_zero(bf16_t* __restrict__ d, int n)
{
    const int i = blockIdx.x * blockDim.x + threadIdx.x;
    if (i < n) d[i] = u16_as_bf(0);
}

// ---------------------------------------------------------------------------
// Global average pool over 49 positions: X [128*49, 2048] -> pooled [128,2048]
// ---------------------------------------------------------------------------
__global__ void k_pool(const bf16_t* __restrict__ X, bf16_t* __restrict__ out)
{
    const int r = blockIdx.x;
    for (int c = threadIdx.x; c < 2048; c += blockDim.x) {
        float s = 0.0f;
        for (int t = 0; t < 49; ++t)
            s += bf2f(X[((size_t)r * 49 + t) * 2048 + c]);
        out[(size_t)r * 2048 + c] = f2bf(s * (1.0f / 49.0f));
    }
}

// ---------------------------------------------------------------------------
// Host-side orchestration
// ---------------------------------------------------------------------------
extern "C" void kernel_launch(void* const* d_in, const int* in_sizes, int n_in,
                              void* d_out, int out_size, void* d_ws, size_t ws_size,
                              hipStream_t stream)
{
    // ---- input index mapping (handles both flattening conventions) ----
    int iFeat = 0, iBox = 0, iBidx = 0;
    int ic1w[3], ic1s[3], ic1b[3], ic2w[3], ic2s[3], ic2b[3];
    int ic3w[3], ic3s[3], ic3b[3];
    int iscw = 0, iscs = 0, iscb = 0, iclsw = 0, iclsb = 0, ibbw = 0, ibbb = 0;

    if (in_sizes[0] == 128) {
        // jax pytree sorted-key order
        iBidx = 0; iBox = 1; iFeat = 2; ibbb = 3; ibbw = 4;
        ic1b[0] = 5;  ic1s[0] = 6;  ic1w[0] = 7;
        ic2b[0] = 8;  ic2s[0] = 9;  ic2w[0] = 10;
        ic3b[0] = 11; ic3s[0] = 12; ic3w[0] = 13;
        iscb = 14; iscs = 15; iscw = 16;
        ic1b[1] = 17; ic1s[1] = 18; ic1w[1] = 19;
        ic2b[1] = 20; ic2s[1] = 21; ic2w[1] = 22;
        ic3b[1] = 23; ic3s[1] = 24; ic3w[1] = 25;
        ic1b[2] = 26; ic1s[2] = 27; ic1w[2] = 28;
        ic2b[2] = 29; ic2s[2] = 30; ic2w[2] = 31;
        ic3b[2] = 32; ic3s[2] = 33; ic3w[2] = 34;
        iclsb = 35; iclsw = 36;
    } else {
        // insertion order
        iFeat = 0; iBox = 1; iBidx = 2;
        int p = 3;
        for (int blk = 0; blk < 3; ++blk) {
            ic1w[blk] = p++; ic1s[blk] = p++; ic1b[blk] = p++;
            ic2w[blk] = p++; ic2s[blk] = p++; ic2b[blk] = p++;
            ic3w[blk] = p++; ic3s[blk] = p++; ic3b[blk] = p++;
            if (blk == 0) { iscw = p++; iscs = p++; iscb = p++; }
        }
        iclsw = p++; iclsb = p++; ibbw = p++; ibbb = p++;
    }

    const float* feat  = (const float*)d_in[iFeat];
    const float* boxes = (const float*)d_in[iBox];
    const int*   bidx  = (const int*)d_in[iBidx];
    auto F = [&](int i) { return (const float*)d_in[i]; };

    // ---- workspace carve ----
    char* base = (char*)d_ws;
    size_t off = 0;
    auto carve = [&](size_t bytes) -> char* {
        char* p = base + off;
        off = (off + bytes + 255) & ~(size_t)255;
        return p;
    };
    bf16_t* X0 = (bf16_t*)carve((size_t)M0 * CIN * 2);
    bf16_t* Wc1[3], *Wc2[3], *Wc3[3];
    Wc1[0] = (bf16_t*)carve((size_t)CMID * CIN * 2);
    Wc2[0] = (bf16_t*)carve((size_t)CMID * 4608 * 2);
    Wc3[0] = (bf16_t*)carve((size_t)COUT * CMID * 2);
    bf16_t* Wsc = (bf16_t*)carve((size_t)COUT * CIN * 2);
    for (int b = 1; b < 3; ++b) {
        Wc1[b] = (bf16_t*)carve((size_t)CMID * COUT * 2);
        Wc2[b] = (bf16_t*)carve((size_t)CMID * 4608 * 2);
        Wc3[b] = (bf16_t*)carve((size_t)COUT * CMID * 2);
    }
    bf16_t* Wcls = (bf16_t*)carve((size_t)NCLS * COUT * 2);
    bf16_t* Wbb  = (bf16_t*)carve((size_t)NBBX * COUT * 2);
    bf16_t* Y1   = (bf16_t*)carve((size_t)M0 * CMID * 2);
    bf16_t* Y2   = (bf16_t*)carve((size_t)M1 * CMID * 2);
    float*  Sres = (float*)carve((size_t)M1 * COUT * 4);
    bf16_t* X1   = (bf16_t*)carve((size_t)M1 * COUT * 2);
    bf16_t* X2   = (bf16_t*)carve((size_t)M1 * COUT * 2);
    bf16_t* pooled = (bf16_t*)carve((size_t)NBOX * COUT * 2);
    bf16_t* zbuf   = (bf16_t*)carve(256);        // zero source for im2col OOB
    (void)ws_size; (void)n_in; (void)out_size;

    // ---- weight conversion + zero buffer ----
    auto cvt = [&](int idx, bf16_t* dst, int count) {
        k_convert<<<dim3((count + 255) / 256), dim3(256), 0, stream>>>(
            F(idx), dst, count);
    };
    k_zero<<<dim3(1), dim3(128), 0, stream>>>(zbuf, 128);
    cvt(ic1w[0], Wc1[0], CMID * CIN);
    k_convert3x3<<<dim3((512 * 4608 + 255) / 256), dim3(256), 0, stream>>>(
        F(ic2w[0]), Wc2[0]);
    cvt(ic3w[0], Wc3[0], COUT * CMID);
    cvt(iscw, Wsc, COUT * CIN);
    for (int b = 1; b < 3; ++b) {
        cvt(ic1w[b], Wc1[b], CMID * COUT);
        k_convert3x3<<<dim3((512 * 4608 + 255) / 256), dim3(256), 0, stream>>>(
            F(ic2w[b]), Wc2[b]);
        cvt(ic3w[b], Wc3[b], COUT * CMID);
    }
    cvt(iclsw, Wcls, NCLS * COUT);
    cvt(ibbw, Wbb, NBBX * COUT);

    // ---- ROI align ----
    k_roi<<<dim3(M0), dim3(256), 0, stream>>>(feat, boxes, bidx, X0);

    const dim3 blk(256);
    // ---- block 0 (stride 2) ----
    k_gemm<AM_PLAIN, EP_BNRELU_BF16><<<dim3(4, 196), blk, 0, stream>>>(
        X0, Wc1[0], F(ic1s[0]), F(ic1b[0]), nullptr, Y1, zbuf,
        CMID, CIN, CMID, 0, 0);
    k_gemm<AM_IM2COL, EP_BNRELU_BF16><<<dim3(4, 49), blk, 0, stream>>>(
        Y1, Wc2[0], F(ic2s[0]), F(ic2b[0]), nullptr, Y2, zbuf,
        CMID, 4608, CMID, 14, 2);
    k_gemm<AM_STRIDE2, EP_BN_F32><<<dim3(16, 49), blk, 0, stream>>>(
        X0, Wsc, F(iscs), F(iscb), nullptr, Sres, zbuf,
        COUT, CIN, COUT, 0, 0);
    k_gemm<AM_PLAIN, EP_BN_ADDF32_RELU_BF16><<<dim3(16, 49), blk, 0, stream>>>(
        Y2, Wc3[0], F(ic3s[0]), F(ic3b[0]), Sres, X1, zbuf,
        COUT, CMID, COUT, 0, 0);

    // ---- block 1 ----
    k_gemm<AM_PLAIN, EP_BNRELU_BF16><<<dim3(4, 49), blk, 0, stream>>>(
        X1, Wc1[1], F(ic1s[1]), F(ic1b[1]), nullptr, Y1, zbuf,
        CMID, COUT, CMID, 0, 0);
    k_gemm<AM_IM2COL, EP_BNRELU_BF16><<<dim3(4, 49), blk, 0, stream>>>(
        Y1, Wc2[1], F(ic2s[1]), F(ic2b[1]), nullptr, Y2, zbuf,
        CMID, 4608, CMID, 7, 1);
    k_gemm<AM_PLAIN, EP_BN_ADDBF16_RELU_BF16><<<dim3(16, 49), blk, 0, stream>>>(
        Y2, Wc3[1], F(ic3s[1]), F(ic3b[1]), X1, X2, zbuf,
        COUT, CMID, COUT, 0, 0);

    // ---- block 2 ----
    k_gemm<AM_PLAIN, EP_BNRELU_BF16><<<dim3(4, 49), blk, 0, stream>>>(
        X2, Wc1[2], F(ic1s[2]), F(ic1b[2]), nullptr, Y1, zbuf,
        CMID, COUT, CMID, 0, 0);
    k_gemm<AM_IM2COL, EP_BNRELU_BF16><<<dim3(4, 49), blk, 0, stream>>>(
        Y1, Wc2[2], F(ic2s[2]), F(ic2b[2]), nullptr, Y2, zbuf,
        CMID, 4608, CMID, 7, 1);
    k_gemm<AM_PLAIN, EP_BN_ADDBF16_RELU_BF16><<<dim3(16, 49), blk, 0, stream>>>(
        Y2, Wc3[2], F(ic3s[2]), F(ic3b[2]), X2, X1, zbuf,
        COUT, CMID, COUT, 0, 0);

    // ---- pool + FC heads (output [128, 8001] f32) ----
    k_pool<<<dim3(NBOX), blk, 0, stream>>>(X1, pooled);
    k_gemm<AM_PLAIN, EP_BIAS_F32><<<dim3(13, 1), blk, 0, stream>>>(
        pooled, Wcls, nullptr, F(iclsb), nullptr, d_out, zbuf,
        NCLS, COUT, NCLS + NBBX, 0, 0);
    k_gemm<AM_PLAIN, EP_BIAS_F32><<<dim3(50, 1), blk, 0, stream>>>(
        pooled, Wbb, nullptr, F(ibbb), nullptr, (float*)d_out + NCLS, zbuf,
        NBBX, COUT, NCLS + NBBX, 0, 0);
}
